// CausalSelfAttention_3418793968326
// MI455X (gfx1250) — matrix operand order repaired and checked by reading, not compiled, not run on hardware
//
#include <hip/hip_runtime.h>
#include <hip/hip_bf16.h>
#include <math.h>

// ---------------------------------------------------------------------------
// Causal self-attention for MI455X (gfx1250), wave32 + WMMA f16 (f32 accum).
//   Stage 1: Q/K/V projections -> f16 in workspace (V transposed).
//   Stage 2: flash-attention (online softmax). 32 q-rows per 8-wave block;
//            each block processes the causal-complementary pair (qb, 63-qb)
//            so all blocks do identical work (~33 key-blocks). Q/K d-chunks
//            are staged with GLOBAL_LOAD_ASYNC_TO_LDS_B128, double-buffered
//            on ASYNCcnt so copies overlap WMMA.
// ---------------------------------------------------------------------------

typedef _Float16 v16h __attribute__((ext_vector_type(16)));
typedef _Float16 v8h  __attribute__((ext_vector_type(8)));
typedef float    v8f  __attribute__((ext_vector_type(8)));

#define INP   1024
#define SEQ   2048
#define NBAT  4
#define NTOK  (NBAT * SEQ)   // 8192 tokens

static __device__ __forceinline__ v8f wmma_f16(v16h a, v16h b, v8f c) {
  // D = A(16x32 f16) * B(32x16 f16) + C(16x16 f32)
  return __builtin_amdgcn_wmma_f32_16x16x32_f16(
      /*neg_a=*/false, a, /*neg_b=*/false, b,
      /*c_mod=*/(short)0, c, /*reuse_a=*/false, /*reuse_b=*/false);
}

// Async 16-byte global->LDS copy, per lane. Tracked by ASYNCcnt.
static __device__ __forceinline__ void async_g2l_b128(unsigned lds_addr,
                                                      unsigned long long gaddr) {
  asm volatile("global_load_async_to_lds_b128 %0, %1, off"
               :: "v"(lds_addr), "v"(gaddr)
               : "memory");
}
#define WAIT_ASYNC_3() asm volatile("s_wait_asynccnt 3" ::: "memory")
#define WAIT_ASYNC_0() asm volatile("s_wait_asynccnt 0" ::: "memory")

// ===========================================================================
// Kernel 1: out = X @ W^T  (nn.Linear), X:[8192,1024] f32, W:[1024,1024] f32.
// 8 waves/block, block tile 128(M) x 64(N), K-step 32. W tile converted to
// f16 once and staged in LDS, shared across the 8 waves. z selects Wq/Wk/Wv.
// Q,K stored row-major f16 [token][d]; V stored transposed f16 [d][token].
// ===========================================================================
__global__ __launch_bounds__(256)
void qkv_proj_kernel(const float* __restrict__ X,
                     const float* __restrict__ Wq,
                     const float* __restrict__ Wk,
                     const float* __restrict__ Wv,
                     _Float16* __restrict__ Qh,
                     _Float16* __restrict__ Kh,
                     _Float16* __restrict__ Vt)
{
  const int z  = blockIdx.z;
  const float* W = (z == 0) ? Wq : ((z == 1) ? Wk : Wv);

  const int m0   = blockIdx.x * 128;
  const int n0   = blockIdx.y * 64;
  const int tid  = threadIdx.x;
  const int wid  = tid >> 5;      // wave id 0..7 -> 16 M-rows each
  const int lane = tid & 31;
  const int l16  = lane & 15;
  const int lhi  = lane >> 4;     // 0 or 1 (half-wave)

  __shared__ __attribute__((aligned(64))) _Float16 Bt[64][32]; // W tile, f16

  v8f acc[4] = {};                // 16x64 output per wave (4 N-tiles)

  const int wrow = tid >> 2;          // 0..63 : W row within tile
  const int wc8  = (tid & 3) * 8;     // 0,8,16,24 : K sub-chunk
  const int arow = m0 + wid * 16 + l16;

  for (int kc = 0; kc < INP; kc += 32) {
    // ---- cooperative stage of W[n0..n0+63][kc..kc+31] as f16 into LDS ----
    {
      const float* src = W + (size_t)(n0 + wrow) * INP + kc + wc8;
      float4 f0 = *(const float4*)(src);
      float4 f1 = *(const float4*)(src + 4);
      v8h h;
      h[0] = (_Float16)f0.x; h[1] = (_Float16)f0.y;
      h[2] = (_Float16)f0.z; h[3] = (_Float16)f0.w;
      h[4] = (_Float16)f1.x; h[5] = (_Float16)f1.y;
      h[6] = (_Float16)f1.z; h[7] = (_Float16)f1.w;
      *(v8h*)&Bt[wrow][wc8] = h;
    }
    __syncthreads();

    // ---- A operand: X rows, contiguous k order (lane m: K 0..15 /
    //      lane m+16: K 16..31), fp32 -> f16 ----
    const float* asrc = X + (size_t)arow * INP + kc + lhi * 16;
    float4 a0 = *(const float4*)(asrc);
    float4 a1 = *(const float4*)(asrc + 4);
    float4 a2 = *(const float4*)(asrc + 8);
    float4 a3 = *(const float4*)(asrc + 12);
    v16h A;
    A[0]  = (_Float16)a0.x; A[1]  = (_Float16)a0.y;
    A[2]  = (_Float16)a0.z; A[3]  = (_Float16)a0.w;
    A[4]  = (_Float16)a1.x; A[5]  = (_Float16)a1.y;
    A[6]  = (_Float16)a1.z; A[7]  = (_Float16)a1.w;
    A[8]  = (_Float16)a2.x; A[9]  = (_Float16)a2.y;
    A[10] = (_Float16)a2.z; A[11] = (_Float16)a2.w;
    A[12] = (_Float16)a3.x; A[13] = (_Float16)a3.y;
    A[14] = (_Float16)a3.z; A[15] = (_Float16)a3.w;

    // ---- B operands from LDS: lane n = output column, 16 consecutive K ----
#pragma unroll
    for (int t = 0; t < 4; ++t) {
      const int c = t * 16 + l16;
      v16h B = *(const v16h*)&Bt[c][lhi * 16];
      acc[t] = wmma_f16(A, B, acc[t]);
    }
    __syncthreads();
  }

  // ---- store D tiles as f16 ----
  if (z < 2) {
    _Float16* outp = (z == 0) ? Qh : Kh;   // row-major [token][d]
#pragma unroll
    for (int t = 0; t < 4; ++t) {
      const int col = n0 + t * 16 + l16;
#pragma unroll
      for (int v = 0; v < 8; ++v) {
        const int row = m0 + wid * 16 + lhi * 8 + v;
        outp[(size_t)row * INP + col] = (_Float16)acc[t][v];
      }
    }
  } else {
    // V transposed: Vt[d][token]; M-dimension is contiguous per lane -> b128
#pragma unroll
    for (int t = 0; t < 4; ++t) {
      const int col  = n0 + t * 16 + l16;
      const int row0 = m0 + wid * 16 + lhi * 8;
      v8h hv;
#pragma unroll
      for (int v = 0; v < 8; ++v) hv[v] = (_Float16)acc[t][v];
      *(v8h*)&Vt[(size_t)col * NTOK + row0] = hv;
    }
  }
}

// ===========================================================================
// Kernel 2: flash attention. Block = 256 threads = 8 waves (2x4 grid):
//   wi = q sub-block (16 rows of 32), wj = output column block (256 cols).
// Each block handles two complementary 32-row q-blocks: (qb, 63-qb) so total
// work per block is constant. Per 64-key block: WMMA scores (32x64) -> LDS,
// fp32 online softmax (1 thread/row), rescale O, WMMA P@V into 16 f32
// accumulators per wave. Q/K chunks double-buffered via async global->LDS.
// ===========================================================================
__global__ __launch_bounds__(256)
void flash_attn_kernel(const _Float16* __restrict__ Qh,
                       const _Float16* __restrict__ Kh,
                       const _Float16* __restrict__ Vt,
                       float* __restrict__ Out)
{
  const int b    = blockIdx.y;    // 0..3  : batch
  const int tid  = threadIdx.x;
  const int wid  = tid >> 5;      // 0..7
  const int lane = tid & 31;
  const int l16  = lane & 15;
  const int lhi  = lane >> 4;
  const int wi   = wid >> 2;      // 0..1 : q sub-block (16 rows)
  const int wj   = wid & 3;       // 0..3 : output column block (256 cols)

  __shared__ __attribute__((aligned(64))) _Float16 Qc[2][32][64];  // ping-pong
  __shared__ __attribute__((aligned(64))) _Float16 Kc[2][64][64];  // ping-pong
  __shared__ __attribute__((aligned(64))) float    Sld[32][64];
  __shared__ __attribute__((aligned(64))) _Float16 Pld[32][64];
  __shared__ float mrow[32];
  __shared__ float lrow[32];
  __shared__ float arow[32];

  // cooperative async-copy mapping: 16B per thread per buffer row-chunk
  const int crow = tid >> 3;        // 0..31
  const int cc8  = (tid & 7) * 8;   // 8-half chunk (16 bytes)

  const unsigned qlds[2] = { (unsigned)(size_t)&Qc[0][crow][cc8],
                             (unsigned)(size_t)&Qc[1][crow][cc8] };
  const unsigned klds0[2] = { (unsigned)(size_t)&Kc[0][crow][cc8],
                              (unsigned)(size_t)&Kc[1][crow][cc8] };
  const unsigned klds1[2] = { (unsigned)(size_t)&Kc[0][32 + crow][cc8],
                              (unsigned)(size_t)&Kc[1][32 + crow][cc8] };

  for (int half = 0; half < 2; ++half) {
    // complementary pairing: work(qb) + work(63-qb) == const
    const int qb    = (half == 0) ? blockIdx.x : 63 - blockIdx.x;  // 0..63
    const int qtok0 = b * SEQ + qb * 32;
    const int nkb   = (qb * 32 + 31) / 64 + 1;   // # of 64-wide key blocks

    v8f o[16] = {};   // 16 q-rows x 256 out-cols, fp32

    __syncthreads();   // LDS quiescent before reuse across halves
    if (tid < 32) { mrow[tid] = -__builtin_inff(); lrow[tid] = 0.0f; }

    const unsigned long long qg0 =
        (unsigned long long)(size_t)(Qh + (size_t)(qtok0 + crow) * INP + cc8);

    for (int kb = 0; kb < nkb; ++kb) {
      const int ktok0 = b * SEQ + kb * 64;

      // ---------------- scores: S = Q . K^T over full d=1024 --------------
      v8f s = {};
      {
        unsigned long long qa = qg0;
        unsigned long long ka0 = (unsigned long long)(size_t)
            (Kh + (size_t)(ktok0 + crow) * INP + cc8);
        unsigned long long ka1 = (unsigned long long)(size_t)
            (Kh + (size_t)(ktok0 + 32 + crow) * INP + cc8);
        // prologue: fill buffer 0 (d-chunk 0): 3 async copies per thread
        async_g2l_b128(qlds[0], qa);
        async_g2l_b128(klds0[0], ka0);
        async_g2l_b128(klds1[0], ka1);
        for (int i = 0; i < 16; ++i) {       // 16 x 64-wide d-chunks
          const int cur = i & 1;
          if (i < 15) {
            qa += 128; ka0 += 128; ka1 += 128;     // next 64 halves
            async_g2l_b128(qlds[cur ^ 1], qa);
            async_g2l_b128(klds0[cur ^ 1], ka0);
            async_g2l_b128(klds1[cur ^ 1], ka1);
            WAIT_ASYNC_3();                  // older triple landed (in-order)
          } else {
            WAIT_ASYNC_0();
          }
          __syncthreads();                   // buf[cur] complete in all waves
#pragma unroll
          for (int dd = 0; dd < 64; dd += 32) {
            v8h qlo = *(const v8h*)&Qc[cur][wi * 16 + l16][dd + lhi * 16];
            v8h qhi = *(const v8h*)&Qc[cur][wi * 16 + l16][dd + lhi * 16 + 8];
            v16h A;
#pragma unroll
            for (int e = 0; e < 8; ++e) { A[e] = qlo[e]; A[e + 8] = qhi[e]; }
            v16h B = *(const v16h*)&Kc[cur][wj * 16 + l16][dd + lhi * 16];
            s = wmma_f16(A, B, s);
          }
          __syncthreads();                   // reads done before buf reuse
        }
      }
#pragma unroll
      for (int v = 0; v < 8; ++v)
        Sld[wi * 16 + lhi * 8 + v][wj * 16 + l16] = s[v];
      __syncthreads();

      // -------------- online softmax (fp32), one thread per q-row ---------
      if (tid < 32) {
        const int   r     = tid;
        const int   qg    = qb * 32 + r;
        const float scale = 0.03125f;               // 1/sqrt(1024)
        const float ninf  = -__builtin_inff();
        float mold = mrow[r];
        float bmax = ninf;
        for (int c = 0; c < 64; ++c) {
          float x = ((kb * 64 + c) <= qg) ? Sld[r][c] * scale : ninf;
          bmax = fmaxf(bmax, x);
        }
        float mnew = fmaxf(mold, bmax);
        float al   = __expf(mold - mnew);           // 0 on first block
        float ssum = 0.0f;
        for (int c = 0; c < 64; ++c) {
          float x = ((kb * 64 + c) <= qg) ? Sld[r][c] * scale : ninf;
          float p = __expf(x - mnew);               // masked -> exp(-inf)=0
          ssum += p;
          Pld[r][c] = (_Float16)p;
        }
        mrow[r] = mnew;
        lrow[r] = lrow[r] * al + ssum;
        arow[r] = al;
      }
      __syncthreads();

      // -------------- O = O*alpha + P @ V ---------------------------------
      float av[8];
#pragma unroll
      for (int v = 0; v < 8; ++v) av[v] = arow[wi * 16 + lhi * 8 + v];
#pragma unroll
      for (int t = 0; t < 16; ++t)
#pragma unroll
        for (int v = 0; v < 8; ++v) o[t][v] *= av[v];

#pragma unroll
      for (int kk = 0; kk < 64; kk += 32) {
        v8h plo = *(const v8h*)&Pld[wi * 16 + l16][kk + lhi * 16];
        v8h phi = *(const v8h*)&Pld[wi * 16 + l16][kk + lhi * 16 + 8];
        v16h A;
#pragma unroll
        for (int e = 0; e < 8; ++e) { A[e] = plo[e]; A[e + 8] = phi[e]; }
        const int vtok = ktok0 + kk + lhi * 16;   // 16 consecutive tokens
#pragma unroll
        for (int t = 0; t < 16; ++t) {
          const int p = wj * 256 + t * 16 + l16;  // output column
          v16h B = *(const v16h*)&Vt[(size_t)p * NTOK + vtok];
          o[t] = wmma_f16(A, B, o[t]);
        }
      }
      __syncthreads();
    }

    // -------------- normalize by 1/l and store fp32 -----------------------
    float lv[8];
#pragma unroll
    for (int v = 0; v < 8; ++v) lv[v] = 1.0f / lrow[wi * 16 + lhi * 8 + v];
#pragma unroll
    for (int t = 0; t < 16; ++t) {
      const int p = wj * 256 + t * 16 + l16;
#pragma unroll
      for (int v = 0; v < 8; ++v) {
        const int row = qtok0 + wi * 16 + lhi * 8 + v;
        Out[(size_t)row * INP + p] = o[t][v] * lv[v];
      }
    }
  }
}

// ===========================================================================
extern "C" void kernel_launch(void* const* d_in, const int* in_sizes, int n_in,
                              void* d_out, int out_size, void* d_ws, size_t ws_size,
                              hipStream_t stream) {
  (void)in_sizes; (void)n_in; (void)out_size; (void)ws_size;
  const float* X  = (const float*)d_in[0];
  const float* Wq = (const float*)d_in[1];
  const float* Wk = (const float*)d_in[2];
  const float* Wv = (const float*)d_in[3];
  float* Out = (float*)d_out;

  // workspace: Q (16MB f16) | K (16MB f16) | V^T (16MB f16)  => 48MB
  _Float16* Qh = (_Float16*)d_ws;
  _Float16* Kh = Qh + (size_t)NTOK * INP;
  _Float16* Vt = Kh + (size_t)NTOK * INP;

  qkv_proj_kernel<<<dim3(NTOK / 128, INP / 64, 3), 256, 0, stream>>>(
      X, Wq, Wk, Wv, Qh, Kh, Vt);
  // 32 q-block pairs x 4 batches; each block does the (qb, 63-qb) pair
  flash_attn_kernel<<<dim3(32, NBAT), 256, 0, stream>>>(
      Qh, Kh, Vt, Out);
}
